// TaskDrivenAttentionModel_32547262169111
// MI455X (gfx1250) — matrix operand order repaired and checked by reading, not compiled, not run on hardware
//
#include <hip/hip_runtime.h>
#include <hip/hip_bf16.h>
#include <math.h>
#include <stdint.h>

// ---------------- model dims ----------------
#define BATCH 512
#define NTOK  300
#define KTOP  20
#define EMB   256
#define FEAT  2048
#define NCLS  1000
#define NTASK 80
#define MROWS (BATCH * KTOP)   // 10240
#define HEADS 4
#define HDIM  64

typedef __bf16 v16bf __attribute__((ext_vector_type(16)));
typedef __bf16 v8bf  __attribute__((ext_vector_type(8)));
typedef float  v8f   __attribute__((ext_vector_type(8)));

__device__ __forceinline__ v8bf cvt8(const float4& a, const float4& b) {
    v8bf r;
    r[0] = (__bf16)a.x; r[1] = (__bf16)a.y; r[2] = (__bf16)a.z; r[3] = (__bf16)a.w;
    r[4] = (__bf16)b.x; r[5] = (__bf16)b.y; r[6] = (__bf16)b.z; r[7] = (__bf16)b.w;
    return r;
}

// =====================================================================
// Top-K (20 of 300) per batch: one wave per batch, iterative wave-argmax.
// Tie-break: lower index wins (matches jax.lax.top_k).
// =====================================================================
__global__ __launch_bounds__(32)
void topk_kernel(const float* __restrict__ scores, const unsigned char* __restrict__ mask,
                 int* __restrict__ absrow, float* __restrict__ dsg, int* __restrict__ maskg) {
    __shared__ float sc[NTOK];
    const int b = blockIdx.x;
    const int lane = threadIdx.x;
    for (int i = lane; i < NTOK; i += 32) sc[i] = scores[b * NTOK + i];
    __syncthreads();
    for (int t = 0; t < KTOP; ++t) {
        float bvv = -3e38f; int bi = 0x7fffffff;
        for (int i = lane; i < NTOK; i += 32) {
            float v = sc[i];
            if (v > bvv || (v == bvv && i < bi)) { bvv = v; bi = i; }
        }
        #pragma unroll
        for (int off = 16; off; off >>= 1) {
            float ov = __shfl_xor(bvv, off, 32);
            int   oi = __shfl_xor(bi, off, 32);
            if (ov > bvv || (ov == bvv && oi < bi)) { bvv = ov; bi = oi; }
        }
        if (lane == 0) {
            absrow[b * KTOP + t] = b * NTOK + bi;
            dsg[b * KTOP + t]    = bvv;
            maskg[b * KTOP + t]  = mask[b * NTOK + bi] ? 1 : 0;
            sc[bi] = -3e38f;
        }
        __syncthreads();
    }
}

// =====================================================================
// Fused WMMA GEMM: out[m,n] = epi( sum_k A[row(m),k] * W[n,k] + bias[n] )
//   Block = 128 threads = 4 waves; tile M=64, N=64, Kstep=32.
//   Each wave computes 32x32: 2 A frags x 2 B frags = 4 wmma per step.
//   Global loads for step k+1 issued before compute(k) (reg pipeline);
//   f32->bf16 conversion folded into LDS-store of the next iteration.
//   NGUARD=false removes all N-bound checks (N multiple of 64).
// Epilogues: 0=relu  1=relu*extra[m,n]  2=*extra[m] (ds scale)
//            3=bias only  4=+extra[m,n] (residual)  5=sigmoid
// =====================================================================
template<int EPI, bool NGUARD>
__global__ __launch_bounds__(128)
void gemm_wmma(const float* __restrict__ A, const int* __restrict__ rowidx,
               const float* __restrict__ W, const float* __restrict__ bias,
               const float* __restrict__ extra, float* __restrict__ out,
               int Ktot, int Nw, int lda) {
    // 40-element (80B) pitch: 16B-aligned fragment loads, conflict-free banks
    __shared__ __align__(16) __bf16 aS[64 * 40];
    __shared__ __align__(16) __bf16 bS[64 * 40];

    const int t    = threadIdx.x;
    const int lane = t & 31;
    const int wave = t >> 5;
    const int wm   = wave >> 1;          // 32-row sub-tile
    const int wn   = wave & 1;           // 32-col sub-tile
    const int mbase = blockIdx.y * 64;
    const int nbase = blockIdx.x * 64;

    v8f acc[2][2];
    #pragma unroll
    for (int i = 0; i < 2; ++i)
        #pragma unroll
        for (int j = 0; j < 2; ++j)
            #pragma unroll
            for (int r = 0; r < 8; ++r) acc[i][j][r] = 0.f;

    // ---- staging assignment: row = t>>1 (0..63), 16-wide K chunk ----
    const int arow = t >> 1;
    const int kq   = (t & 1) * 16;
    const int srow = rowidx ? rowidx[mbase + arow] : (mbase + arow);
    const float* asrc = A + (size_t)srow * (size_t)lda + kq;
    const int nn   = nbase + arow;
    const bool bval = NGUARD ? (nn < Nw) : true;
    const float* wsrc = W + (size_t)(bval ? nn : 0) * (size_t)Ktot + kq;

    // fragment addressing (constant across K loop)
    const int am0 = wm * 32 + (lane & 15);
    const int akb = (lane < 16) ? 0 : 16;
    const int bn0 = wn * 32 + (lane & 15);
    const int bkb = (lane < 16) ? 0 : 16;

    const int Kmain = Ktot & ~31;

    float4 ra[4], rb[4];
    auto gload = [&](int kc) {
        ra[0] = *(const float4*)(asrc + kc);
        ra[1] = *(const float4*)(asrc + kc + 4);
        ra[2] = *(const float4*)(asrc + kc + 8);
        ra[3] = *(const float4*)(asrc + kc + 12);
        if (bval) {
            rb[0] = *(const float4*)(wsrc + kc);
            rb[1] = *(const float4*)(wsrc + kc + 4);
            rb[2] = *(const float4*)(wsrc + kc + 8);
            rb[3] = *(const float4*)(wsrc + kc + 12);
        } else {
            rb[0] = rb[1] = rb[2] = rb[3] = make_float4(0.f, 0.f, 0.f, 0.f);
        }
    };
    auto sstore = [&]() {
        *(v8bf*)(&aS[arow * 40 + kq])     = cvt8(ra[0], ra[1]);
        *(v8bf*)(&aS[arow * 40 + kq + 8]) = cvt8(ra[2], ra[3]);
        *(v8bf*)(&bS[arow * 40 + kq])     = cvt8(rb[0], rb[1]);
        *(v8bf*)(&bS[arow * 40 + kq + 8]) = cvt8(rb[2], rb[3]);
    };
    auto compute = [&]() {
        union { v16bf v; v8bf h[2]; } af0, af1, bf0, bf1;
        af0.h[0] = *(const v8bf*)(&aS[am0 * 40 + akb]);
        af0.h[1] = *(const v8bf*)(&aS[am0 * 40 + akb + 8]);
        af1.h[0] = *(const v8bf*)(&aS[(am0 + 16) * 40 + akb]);
        af1.h[1] = *(const v8bf*)(&aS[(am0 + 16) * 40 + akb + 8]);
        bf0.h[0] = *(const v8bf*)(&bS[bn0 * 40 + bkb]);
        bf0.h[1] = *(const v8bf*)(&bS[bn0 * 40 + bkb + 8]);
        bf1.h[0] = *(const v8bf*)(&bS[(bn0 + 16) * 40 + bkb]);
        bf1.h[1] = *(const v8bf*)(&bS[(bn0 + 16) * 40 + bkb + 8]);
        acc[0][0] = __builtin_amdgcn_wmma_f32_16x16x32_bf16(false, af0.v, false, bf0.v,
                                                            (short)0, acc[0][0], false, false);
        acc[0][1] = __builtin_amdgcn_wmma_f32_16x16x32_bf16(false, af0.v, false, bf1.v,
                                                            (short)0, acc[0][1], false, false);
        acc[1][0] = __builtin_amdgcn_wmma_f32_16x16x32_bf16(false, af1.v, false, bf0.v,
                                                            (short)0, acc[1][0], false, false);
        acc[1][1] = __builtin_amdgcn_wmma_f32_16x16x32_bf16(false, af1.v, false, bf1.v,
                                                            (short)0, acc[1][1], false, false);
    };

    // ---- pipelined main loop ----
    if (Kmain > 0) {
        gload(0);
        for (int kc = 0; kc < Kmain; kc += 32) {
            sstore();
            __syncthreads();
            if (kc + 32 < Kmain) gload(kc + 32);   // overlap with wmma below
            compute();
            __syncthreads();
        }
    }

    // ---- single guarded tail step (e.g. K=1000) ----
    if (Kmain < Ktot) {
        v8bf a0v, a1v, b0v, b1v;
        #pragma unroll
        for (int j = 0; j < 8; ++j) {
            const int k0 = Kmain + kq + j;
            const int k1 = k0 + 8;
            a0v[j] = (__bf16)((k0 < Ktot) ? asrc[Kmain + j] : 0.f);
            a1v[j] = (__bf16)((k1 < Ktot) ? asrc[Kmain + 8 + j] : 0.f);
            b0v[j] = (__bf16)((bval && k0 < Ktot) ? wsrc[Kmain + j] : 0.f);
            b1v[j] = (__bf16)((bval && k1 < Ktot) ? wsrc[Kmain + 8 + j] : 0.f);
        }
        *(v8bf*)(&aS[arow * 40 + kq])     = a0v;
        *(v8bf*)(&aS[arow * 40 + kq + 8]) = a1v;
        *(v8bf*)(&bS[arow * 40 + kq])     = b0v;
        *(v8bf*)(&bS[arow * 40 + kq + 8]) = b1v;
        __syncthreads();
        compute();
    }

    // ---- epilogue ----
    #pragma unroll
    for (int im = 0; im < 2; ++im) {
        #pragma unroll
        for (int in = 0; in < 2; ++in) {
            const int col = nbase + wn * 32 + in * 16 + (lane & 15);
            if (!NGUARD || col < Nw) {
                const float bcol = bias ? bias[col] : 0.f;
                #pragma unroll
                for (int r = 0; r < 8; ++r) {
                    const int m = mbase + wm * 32 + im * 16 + ((lane < 16) ? r : (r + 8));
                    const float a = acc[im][in][r] + bcol;
                    float val;
                    if      (EPI == 0) val = fmaxf(a, 0.f);
                    else if (EPI == 1) val = fmaxf(a, 0.f) * extra[(size_t)m * Nw + col];
                    else if (EPI == 2) val = a * extra[m];
                    else if (EPI == 3) val = a;
                    else if (EPI == 4) val = a + extra[(size_t)m * Nw + col];
                    else               val = 1.f / (1.f + __expf(-a));
                    out[(size_t)m * Nw + col] = val;
                }
            }
        }
    }
}

// =====================================================================
// Attention: one wave per (batch, head). 20x20 scores, masked softmax,
// 20x64 output. Tiny (0.2 GFLOP total) -> plain VALU.
// =====================================================================
__global__ __launch_bounds__(32)
void attn_kernel(const float* __restrict__ qkv, const int* __restrict__ maskg,
                 float* __restrict__ ao) {
    __shared__ float qs[KTOP * HDIM], ks[KTOP * HDIM], vs[KTOP * HDIM], sc[KTOP * KTOP];
    const int b = blockIdx.x >> 2;
    const int h = blockIdx.x & 3;
    const int tid = threadIdx.x;
    const float* base = qkv + (size_t)b * KTOP * (3 * EMB) + h * HDIM;
    for (int i = tid; i < KTOP * HDIM; i += 32) {
        const int s = i >> 6, d = i & 63;
        const float* r = base + (size_t)s * (3 * EMB) + d;
        qs[i] = r[0]; ks[i] = r[EMB]; vs[i] = r[2 * EMB];
    }
    __syncthreads();
    for (int p = tid; p < KTOP * KTOP; p += 32) {
        const int qi = p / KTOP, kj = p % KTOP;
        float dsum = 0.f;
        #pragma unroll
        for (int d = 0; d < HDIM; ++d) dsum += qs[qi * HDIM + d] * ks[kj * HDIM + d];
        dsum *= 0.125f;                        // 1/sqrt(64)
        if (maskg[b * KTOP + kj]) dsum = -1e30f;
        sc[p] = dsum;
    }
    __syncthreads();
    if (tid < KTOP) {
        float mx = -3e38f;
        for (int j = 0; j < KTOP; ++j) mx = fmaxf(mx, sc[tid * KTOP + j]);
        float sum = 0.f;
        for (int j = 0; j < KTOP; ++j) { float e = __expf(sc[tid * KTOP + j] - mx); sc[tid * KTOP + j] = e; sum += e; }
        const float inv = 1.f / sum;
        for (int j = 0; j < KTOP; ++j) sc[tid * KTOP + j] *= inv;
    }
    __syncthreads();
    for (int i = tid; i < KTOP * HDIM; i += 32) {
        const int qi = i >> 6, d = i & 63;
        float o = 0.f;
        #pragma unroll
        for (int j = 0; j < KTOP; ++j) o += sc[qi * KTOP + j] * vs[j * HDIM + d];
        ao[((size_t)b * KTOP + qi) * EMB + h * HDIM + d] = o;
    }
}

// =====================================================================
// LayerNorm over 256 features, one wave per row, shuffle reductions.
// =====================================================================
__global__ __launch_bounds__(32)
void ln_kernel(const float* __restrict__ y, const float* __restrict__ g,
               const float* __restrict__ bta, float* __restrict__ hctx) {
    const int row = blockIdx.x;
    const int lane = threadIdx.x;
    const float* r = y + (size_t)row * EMB;
    float x[8]; float s = 0.f;
    #pragma unroll
    for (int j = 0; j < 8; ++j) { x[j] = r[lane + 32 * j]; s += x[j]; }
    #pragma unroll
    for (int off = 16; off; off >>= 1) s += __shfl_xor(s, off, 32);
    const float mu = s * (1.f / EMB);
    float v = 0.f;
    #pragma unroll
    for (int j = 0; j < 8; ++j) { const float d = x[j] - mu; v += d * d; }
    #pragma unroll
    for (int off = 16; off; off >>= 1) v += __shfl_xor(v, off, 32);
    const float inv = rsqrtf(v * (1.f / EMB) + 1e-5f);
    #pragma unroll
    for (int j = 0; j < 8; ++j) {
        const int e = lane + 32 * j;
        hctx[(size_t)row * EMB + e] = (x[j] - mu) * inv * g[e] + bta[e];
    }
}

__global__ __launch_bounds__(256)
void pfinal_kernel(const float* __restrict__ pc, const float* __restrict__ pv,
                   float* __restrict__ pf, int n) {
    const int i = blockIdx.x * 256 + threadIdx.x;
    if (i < n) pf[i] = 0.5f * (pc[i] + pv[i]);
}

// =====================================================================
extern "C" void kernel_launch(void* const* d_in, const int* in_sizes, int n_in,
                              void* d_out, int out_size, void* d_ws, size_t ws_size,
                              hipStream_t stream) {
    const float* class_onehot = (const float*)d_in[0];
    const float* visual_feats = (const float*)d_in[1];
    const float* det_scores   = (const float*)d_in[2];
    const unsigned char* mask = (const unsigned char*)d_in[3];
    const float* Wc   = (const float*)d_in[4];
    const float* bc   = (const float*)d_in[5];
    const float* Wv   = (const float*)d_in[6];
    const float* bv   = (const float*)d_in[7];
    const float* Wm   = (const float*)d_in[8];
    const float* bm   = (const float*)d_in[9];
    const float* in_w = (const float*)d_in[10];
    const float* in_b = (const float*)d_in[11];
    const float* out_w= (const float*)d_in[12];
    const float* out_b= (const float*)d_in[13];
    const float* ln_g = (const float*)d_in[14];
    const float* ln_b = (const float*)d_in[15];
    const float* Wctx = (const float*)d_in[16];
    const float* bctx = (const float*)d_in[17];
    const float* Wvis = (const float*)d_in[18];
    const float* bvis = (const float*)d_in[19];

    // workspace carve-up (256B aligned)
    size_t off = 0;
    auto take = [&](size_t bytes) -> void* {
        void* p = (void*)((char*)d_ws + off);
        off += (bytes + 255) & ~(size_t)255;
        return p;
    };
    int*   absrow = (int*)  take((size_t)MROWS * 4);
    float* dsg    = (float*)take((size_t)MROWS * 4);
    int*   maskg  = (int*)  take((size_t)MROWS * 4);
    float* ce     = (float*)take((size_t)MROWS * EMB * 4);       // relu(class emb)
    float* hin    = (float*)take((size_t)MROWS * EMB * 4);       // class*visual gate
    float* h0     = (float*)take((size_t)MROWS * EMB * 4);       // combine * ds
    float* qkv    = (float*)take((size_t)MROWS * 3 * EMB * 4);
    float* ao     = (float*)take((size_t)MROWS * EMB * 4);       // attention out
    float* yres   = (float*)take((size_t)MROWS * EMB * 4);       // h0 + out-proj
    float* hctx   = (float*)take((size_t)MROWS * EMB * 4);       // layernorm out

    const int PLANE = BATCH * KTOP * NTASK;                      // 819200
    float* p_final = (float*)d_out;
    float* p_ctx   = p_final + PLANE;
    float* p_vis   = p_ctx + PLANE;

    const dim3 blk(128);
    const int mt = MROWS / 64;                                   // 160

    topk_kernel<<<BATCH, 32, 0, stream>>>(det_scores, mask, absrow, dsg, maskg);

    // class_emb = relu(gather(onehot) @ Wc^T + bc)
    gemm_wmma<0, false><<<dim3(EMB / 64, mt), blk, 0, stream>>>(
        class_onehot, absrow, Wc, bc, nullptr, ce, NCLS, EMB, NCLS);
    // hin = relu(gather(vf) @ Wv^T + bv) * class_emb
    gemm_wmma<1, false><<<dim3(EMB / 64, mt), blk, 0, stream>>>(
        visual_feats, absrow, Wv, bv, ce, hin, FEAT, EMB, FEAT);
    // h0 = (hin @ Wm^T + bm) * ds
    gemm_wmma<2, false><<<dim3(EMB / 64, mt), blk, 0, stream>>>(
        hin, nullptr, Wm, bm, dsg, h0, EMB, EMB, EMB);
    // qkv = h0 @ in_w^T + in_b
    gemm_wmma<3, false><<<dim3((3 * EMB) / 64, mt), blk, 0, stream>>>(
        h0, nullptr, in_w, in_b, nullptr, qkv, EMB, 3 * EMB, EMB);

    attn_kernel<<<BATCH * HEADS, 32, 0, stream>>>(qkv, maskg, ao);

    // y = h0 + (ao @ out_w^T + out_b)
    gemm_wmma<4, false><<<dim3(EMB / 64, mt), blk, 0, stream>>>(
        ao, nullptr, out_w, out_b, h0, yres, EMB, EMB, EMB);

    ln_kernel<<<MROWS, 32, 0, stream>>>(yres, ln_g, ln_b, hctx);

    // p_context = sigmoid(hctx @ Wctx^T + bctx)
    gemm_wmma<5, true><<<dim3((NTASK + 63) / 64, mt), blk, 0, stream>>>(
        hctx, nullptr, Wctx, bctx, nullptr, p_ctx, EMB, NTASK, EMB);
    // p_visual = sigmoid(gather(vf) @ Wvis^T + bvis)
    gemm_wmma<5, true><<<dim3((NTASK + 63) / 64, mt), blk, 0, stream>>>(
        visual_feats, absrow, Wvis, bvis, nullptr, p_vis, FEAT, NTASK, FEAT);

    pfinal_kernel<<<(PLANE + 255) / 256, 256, 0, stream>>>(p_ctx, p_vis, p_final, PLANE);
}
